// MultiHeadedAttentionWithRelativePosition_21345987461680
// MI455X (gfx1250) — matrix operand order repaired and checked by reading, not compiled, not run on hardware
//
#include <hip/hip_runtime.h>

typedef __bf16 bf16;
typedef __attribute__((ext_vector_type(16))) bf16  v16bf;
typedef __attribute__((ext_vector_type(8)))  bf16  v8bf;
typedef __attribute__((ext_vector_type(8)))  float v8f;
typedef __attribute__((ext_vector_type(4)))  unsigned int u32x4;
typedef __attribute__((ext_vector_type(8)))  int i32x8;
typedef __attribute__((ext_vector_type(4)))  int i32x4;

constexpr int CB = 2, CS = 2048, CD = 1024, CH = 16, CDK = 64;
constexpr int CM = CB * CS;              // 4096 rows (B*S)

__device__ __forceinline__ v8f vzero8() {
    v8f v;
#pragma unroll
    for (int i = 0; i < 8; ++i) v[i] = 0.0f;
    return v;
}

__device__ __forceinline__ v16bf make_a_frag(const bf16* p0, const bf16* p1) {
    // A-matrix 16x32 bf16 layout: lane holds two contiguous 8-element chunks:
    //   K = koff..koff+7  (VGPR0-3)  and  K = koff+16..koff+23 (VGPR4-7)
    v8bf lo = *(const v8bf*)p0;
    v8bf hi = *(const v8bf*)p1;
    v16bf a;
#pragma unroll
    for (int i = 0; i < 8; ++i) { a[i] = lo[i]; a[i + 8] = hi[i]; }
    return a;
}

__device__ __forceinline__ float rmax16(float v) {
#pragma unroll
    for (int m = 8; m >= 1; m >>= 1) v = fmaxf(v, __shfl_xor(v, m, 16));
    return v;
}
__device__ __forceinline__ float rsum16(float v) {
#pragma unroll
    for (int m = 8; m >= 1; m >>= 1) v += __shfl_xor(v, m, 16);
    return v;
}

// TDM: DMA a [tile_d1 x tile_d0] bf16 tile (row stride = stride0 elements) from
// global memory into LDS at byte offset lds_off. 2D descriptor, groups 2/3 zero.
__device__ __forceinline__ void tdm_load_tile_bf16(unsigned lds_off, const bf16* gaddr,
                                                   unsigned tensor_d0, unsigned tensor_d1,
                                                   unsigned tile_d0, unsigned tile_d1,
                                                   unsigned stride0) {
    unsigned long long ga = (unsigned long long)(uintptr_t)gaddr;
    u32x4 g0;
    g0[0] = 1u;                                    // count=1, user descriptor
    g0[1] = lds_off;                               // lds_addr (bytes)
    g0[2] = (unsigned)(ga & 0xffffffffu);          // global_addr[31:0]
    g0[3] = (unsigned)((ga >> 32) & 0x01ffffffu)   // global_addr[56:32]
            | 0x80000000u;                         // type=2 ("image")
    i32x8 g1;
    g1[0] = (int)(1u << 16);                       // data_size=1 -> 2 bytes/elem
    g1[1] = (int)((tensor_d0 & 0xffffu) << 16);    // tensor_dim0[15:0] -> bits 63:48
    g1[2] = (int)(((tensor_d0 >> 16) & 0xffffu) | ((tensor_d1 & 0xffffu) << 16));
    g1[3] = (int)(((tensor_d1 >> 16) & 0xffffu) | ((tile_d0 & 0xffffu) << 16));
    g1[4] = (int)(tile_d1 & 0xffffu);              // tile_dim1; tile_dim2=0
    g1[5] = (int)stride0;                          // tensor_dim0_stride[31:0]
    g1[6] = 0;                                     // stride0 hi / dim1_stride lo
    g1[7] = 0;
    i32x4 g2 = {0, 0, 0, 0};
    i32x4 g3 = {0, 0, 0, 0};
#if __clang_major__ >= 23
    i32x8 g4 = {0, 0, 0, 0, 0, 0, 0, 0};
    __builtin_amdgcn_tensor_load_to_lds(g0, g1, g2, g3, g4, 0);
#else
    __builtin_amdgcn_tensor_load_to_lds(g0, g1, g2, g3, 0);
#endif
}

// ---------------------------------------------------------------------------
// Elementwise f32 -> bf16
__global__ void __launch_bounds__(256) k_cvt_bf16(const float* __restrict__ s,
                                                  bf16* __restrict__ d, int n) {
    int i = blockIdx.x * blockDim.x + threadIdx.x;
    if (i < n) d[i] = (bf16)s[i];
}

// Transpose + convert: WT[k*N + n] = W[n*K + k]   (W is [N,K] row-major)
__global__ void __launch_bounds__(256) k_transpose_cvt(const float* __restrict__ W,
                                                       bf16* __restrict__ WT,
                                                       int N, int K) {
    size_t i = (size_t)blockIdx.x * blockDim.x + threadIdx.x;
    if (i >= (size_t)N * K) return;
    int k = (int)(i / N);
    int n = (int)(i % N);
    WT[i] = (bf16)W[(size_t)n * K + k];
}

// ---------------------------------------------------------------------------
// C = X(bf16,[M,K]) @ WT(bf16,[K,N]) + bias.
// Block (8 waves) computes a 128x64 C tile; each wave one 16x64 slice.
// B tiles (64x64 bf16, 8KB) streamed into LDS by the Tensor Data Mover,
// double-buffered (TENSORcnt), shared by all 8 waves; 8 WMMAs per barrier pair.
// mode 0: f32 [M,N] out; mode 1: bf16 [B,H,S,DK]; mode 2: bf16 [B,H,DK,S].
__global__ void __launch_bounds__(256) k_gemm_wmma(const bf16* __restrict__ X,
                                                   const bf16* __restrict__ WT,
                                                   const float* __restrict__ bias,
                                                   void* __restrict__ dst,
                                                   int M, int N, int K, int mode) {
    __shared__ __align__(32) bf16 Bbuf[2][64 * 64];   // double-buffered B tile

    int wid  = threadIdx.x >> 5;
    int lane = threadIdx.x & 31;
    int nb   = N >> 6;
    int bm   = blockIdx.x / nb;
    int n0   = (blockIdx.x % nb) << 6;
    int m0   = bm * 128 + wid * 16;
    int mlo  = lane & 15;
    int koff = (lane >> 4) << 3;             // 0 or 8 per A layout
    int mtop = koff;

    v8f acc[4];
#pragma unroll
    for (int j = 0; j < 4; ++j) acc[j] = vzero8();

    const bf16* arow = X + (size_t)(m0 + mlo) * K;

    if (wid == 0) {
        tdm_load_tile_bf16((unsigned)(uintptr_t)&Bbuf[0][0], WT + n0,
                           (unsigned)N, (unsigned)K, 64u, 64u, (unsigned)N);
    }

    for (int kk = 0; kk < K; kk += 64) {
        int cur = (kk >> 6) & 1;
        if (wid == 0) {
            if (kk + 64 < K) {
                tdm_load_tile_bf16((unsigned)(uintptr_t)&Bbuf[cur ^ 1][0],
                                   WT + (size_t)(kk + 64) * N + n0,
                                   (unsigned)N, (unsigned)K, 64u, 64u, (unsigned)N);
                __builtin_amdgcn_s_wait_tensorcnt(1);   // tile `cur` complete
            } else {
                __builtin_amdgcn_s_wait_tensorcnt(0);
            }
        }
        __syncthreads();                                 // tile `cur` visible to all

#pragma unroll
        for (int dd = 0; dd < 2; ++dd) {
            const bf16* ap0 = arow + kk + dd * 32 + koff;
            v16bf a = make_a_frag(ap0, ap0 + 16);
            const bf16* brow = &Bbuf[cur][(dd * 32 + koff) * 64 + mlo];
#pragma unroll
            for (int j = 0; j < 4; ++j) {
                v16bf b;
#pragma unroll
                for (int s = 0; s < 16; ++s) b[s] = brow[((s & 7) + ((s >> 3) << 4)) * 64 + j * 16];
                acc[j] = __builtin_amdgcn_wmma_f32_16x16x32_bf16(
                    false, a, false, b, (short)0, acc[j], false, false);
            }
        }
        __syncthreads();                                 // done reading before overwrite
    }

#pragma unroll
    for (int j = 0; j < 4; ++j) {
        int n = n0 + j * 16 + mlo;
        float bv = bias[n];
#pragma unroll
        for (int r = 0; r < 8; ++r) {
            int m = m0 + r + mtop;
            float val = acc[j][r] + bv;
            if (mode == 0) {
                ((float*)dst)[(size_t)m * N + n] = val;
            } else if (mode == 1) {
                int b_ = m / CS, s = m % CS, h = n >> 6, dk = n & 63;
                ((bf16*)dst)[((((size_t)b_ * CH + h) * CS + s) * CDK) + dk] = (bf16)val;
            } else {
                int b_ = m / CS, s = m % CS, h = n >> 6, dk = n & 63;
                ((bf16*)dst)[((((size_t)b_ * CH + h) * CDK + dk) * CS) + s] = (bf16)val;
            }
        }
    }
}

// ---------------------------------------------------------------------------
// Flash attention. Block (8 waves) owns (b, h, 128-query stripe); each wave a
// 16-query tile. Shared K-tile (64x32 from K^T) and V-tile (32x64) are streamed
// into LDS by TDM, double-buffered; all B-fragments come from LDS.
__global__ void __launch_bounds__(256) k_attn_wmma(const bf16* __restrict__ Qh,
                                                   const bf16* __restrict__ KT,
                                                   const bf16* __restrict__ Vh,
                                                   const int*  __restrict__ mask,
                                                   bf16* __restrict__ ctx) {
    __shared__ __align__(32) bf16 Kb[2][64 * 32];     // [dk][key] 4KB each
    __shared__ __align__(32) bf16 Vb[2][32 * 64];     // [key][dk] 4KB each
    __shared__ __align__(32) bf16 pbuf[8][16 * 32];   // per-wave P tile

    int wid  = threadIdx.x >> 5;
    int lane = threadIdx.x & 31;
    int qb   = blockIdx.x & 15;             // 16 query stripes of 128
    int h    = (blockIdx.x >> 4) & 15;
    int b    = blockIdx.x >> 8;
    int q0   = qb * 128 + wid * 16;
    int mlo  = lane & 15;
    int khi  = (lane >> 4) << 3;            // 0 or 8
    int mtop = khi;

    const size_t kt_base = ((size_t)b * CH + h) * CDK * (size_t)CS; // + dk*S + s
    const size_t v_base  = ((size_t)b * CH + h) * CS * (size_t)CDK; // + s*DK + dk
    const bf16* kt_tile = KT + kt_base;
    const bf16* v_tile  = Vh + v_base;

    // Q A-frags: dk-steps 0..31 and 32..63
    const bf16* qrow = Qh + (((size_t)b * CH + h) * CS + q0 + mlo) * CDK;
    v16bf aq0 = make_a_frag(qrow + khi,      qrow + khi + 16);
    v16bf aq1 = make_a_frag(qrow + 32 + khi, qrow + 48 + khi);

    v8f o[4];
#pragma unroll
    for (int j = 0; j < 4; ++j) o[j] = vzero8();
    float mrow[8], lrow[8];
#pragma unroll
    for (int r = 0; r < 8; ++r) { mrow[r] = -3.0e38f; lrow[r] = 0.0f; }

    const int* mbase = mask + ((size_t)b * CS + q0) * CS;
    bf16* pw = pbuf[wid];

    if (wid == 0) {
        tdm_load_tile_bf16((unsigned)(uintptr_t)&Kb[0][0], kt_tile,
                           (unsigned)CS, 64u, 32u, 64u, (unsigned)CS);
        tdm_load_tile_bf16((unsigned)(uintptr_t)&Vb[0][0], v_tile,
                           64u, (unsigned)CS, 64u, 32u, 64u);
    }

    for (int sb = 0; sb < CS; sb += 32) {
        int cur = (sb >> 5) & 1;
        if (wid == 0) {
            if (sb + 32 < CS) {
                tdm_load_tile_bf16((unsigned)(uintptr_t)&Kb[cur ^ 1][0],
                                   kt_tile + (sb + 32),
                                   (unsigned)CS, 64u, 32u, 64u, (unsigned)CS);
                tdm_load_tile_bf16((unsigned)(uintptr_t)&Vb[cur ^ 1][0],
                                   v_tile + (size_t)(sb + 32) * CDK,
                                   64u, (unsigned)CS, 64u, 32u, 64u);
                __builtin_amdgcn_s_wait_tensorcnt(2);   // current K+V tiles done
            } else {
                __builtin_amdgcn_s_wait_tensorcnt(0);
            }
        }
        __syncthreads();

        // scores: [16 q] x [32 keys], K-dim = DK=64 in two 32-steps (K from LDS)
        v8f s0 = vzero8(), s1 = vzero8();
#pragma unroll
        for (int dd = 0; dd < 2; ++dd) {
            const bf16* kp = &Kb[cur][(dd * 32 + khi) * 32 + mlo];
            v16bf b0, b1;
#pragma unroll
            for (int s = 0; s < 16; ++s) {
                b0[s] = kp[((s & 7) + ((s >> 3) << 4)) * 32];
                b1[s] = kp[((s & 7) + ((s >> 3) << 4)) * 32 + 16];
            }
            v16bf aq = dd ? aq1 : aq0;
            s0 = __builtin_amdgcn_wmma_f32_16x16x32_bf16(false, aq, false, b0,
                                                         (short)0, s0, false, false);
            s1 = __builtin_amdgcn_wmma_f32_16x16x32_bf16(false, aq, false, b1,
                                                         (short)0, s1, false, false);
        }

        // mask + online softmax (row stats via 16-lane shuffle reductions)
#pragma unroll
        for (int r = 0; r < 8; ++r) {
            const int* mp = mbase + (size_t)(r + mtop) * CS + sb + mlo;
            float e0 = s0[r] * 0.125f;      // 1/sqrt(64)
            float e1 = s1[r] * 0.125f;
            if (mp[0]  == 0) e0 = -1.0e9f;
            if (mp[16] == 0) e1 = -1.0e9f;
            float mx   = rmax16(fmaxf(e0, e1));
            float mnew = fmaxf(mrow[r], mx);
            float corr = __expf(mrow[r] - mnew);
            mrow[r] = mnew;
            e0 = __expf(e0 - mnew);
            e1 = __expf(e1 - mnew);
            lrow[r] = lrow[r] * corr + rsum16(e0 + e1);
            o[0][r] *= corr; o[1][r] *= corr; o[2][r] *= corr; o[3][r] *= corr;
            pw[(r + mtop) * 32 + mlo]      = (bf16)e0;
            pw[(r + mtop) * 32 + 16 + mlo] = (bf16)e1;
        }

        // P (16x32) @ V (32x64): A-frag from LDS P, B-frags from LDS V tile
        const bf16* pp = pw + mlo * 32 + khi;
        v16bf ap = make_a_frag(pp, pp + 16);
#pragma unroll
        for (int j = 0; j < 4; ++j) {
            v16bf bv;
#pragma unroll
            for (int s = 0; s < 16; ++s) bv[s] = Vb[cur][(khi + (s & 7) + ((s >> 3) << 4)) * 64 + j * 16 + mlo];
            o[j] = __builtin_amdgcn_wmma_f32_16x16x32_bf16(false, ap, false, bv,
                                                           (short)0, o[j], false, false);
        }
        __syncthreads();
    }

    // normalize + write context in [B,S,D] bf16
#pragma unroll
    for (int r = 0; r < 8; ++r) {
        float inv = lrow[r] > 0.0f ? 1.0f / lrow[r] : 0.0f;
        int row = q0 + r + mtop;
#pragma unroll
        for (int j = 0; j < 4; ++j) {
            int col = h * CDK + j * 16 + mlo;
            ctx[((size_t)b * CS + row) * CD + col] = (bf16)(o[j][r] * inv);
        }
    }
}

// ---------------------------------------------------------------------------
extern "C" void kernel_launch(void* const* d_in, const int* in_sizes, int n_in,
                              void* d_out, int out_size, void* d_ws, size_t ws_size,
                              hipStream_t stream) {
    const float* query = (const float*)d_in[0];
    const float* key_  = (const float*)d_in[1];
    const float* value = (const float*)d_in[2];
    const int*   mask  = (const int*)d_in[3];
    const float* Wq = (const float*)d_in[4];
    const float* bq = (const float*)d_in[5];
    const float* Wk = (const float*)d_in[6];
    const float* bk = (const float*)d_in[7];
    const float* Wv = (const float*)d_in[8];
    const float* bv = (const float*)d_in[9];
    const float* Wo = (const float*)d_in[10];
    const float* bo = (const float*)d_in[11];

    const size_t MD = (size_t)CM * CD;   // 4 Mi elems
    const size_t DD = (size_t)CD * CD;   // 1 Mi elems

    bf16* p   = (bf16*)d_ws;
    bf16* Xq  = p; p += MD;
    bf16* Xk  = p; p += MD;
    bf16* Xv  = p; p += MD;
    bf16* WqT = p; p += DD;
    bf16* WkT = p; p += DD;
    bf16* WvT = p; p += DD;
    bf16* WoT = p; p += DD;
    bf16* Qh  = p; p += MD;
    bf16* KTt = p; p += MD;
    bf16* Vh  = p; p += MD;
    bf16* ctx = p; p += MD;

    const int thr = 256;
    int nMD = (int)MD, nDD = (int)DD;

    k_cvt_bf16<<<(nMD + thr - 1) / thr, thr, 0, stream>>>(query, Xq, nMD);
    k_cvt_bf16<<<(nMD + thr - 1) / thr, thr, 0, stream>>>(key_,  Xk, nMD);
    k_cvt_bf16<<<(nMD + thr - 1) / thr, thr, 0, stream>>>(value, Xv, nMD);
    k_transpose_cvt<<<(nDD + thr - 1) / thr, thr, 0, stream>>>(Wq, WqT, CD, CD);
    k_transpose_cvt<<<(nDD + thr - 1) / thr, thr, 0, stream>>>(Wk, WkT, CD, CD);
    k_transpose_cvt<<<(nDD + thr - 1) / thr, thr, 0, stream>>>(Wv, WvT, CD, CD);
    k_transpose_cvt<<<(nDD + thr - 1) / thr, thr, 0, stream>>>(Wo, WoT, CD, CD);

    // (M/128)*(D/64) = 32*16 = 512 blocks of 8 waves (128x64 C tile per block)
    k_gemm_wmma<<<512, 256, 0, stream>>>(Xq, WqT, bq, Qh,  CM, CD, CD, 1);
    k_gemm_wmma<<<512, 256, 0, stream>>>(Xk, WkT, bk, KTt, CM, CD, CD, 2);
    k_gemm_wmma<<<512, 256, 0, stream>>>(Xv, WvT, bv, Vh,  CM, CD, CD, 1);

    // B*H*(S/128) = 2*16*16 = 512 blocks; block shares K/V tiles via TDM+LDS
    k_attn_wmma<<<512, 256, 0, stream>>>(Qh, KTt, Vh, mask, ctx);

    k_gemm_wmma<<<512, 256, 0, stream>>>(ctx, WoT, bo, d_out, CM, CD, CD, 0);
}
